// ConCare_35854386987641
// MI455X (gfx1250) — compile-verified, matrix operand order checked by reading, not run on hardware
//
#include <hip/hip_runtime.h>

// ---------------------------------------------------------------------------
// ConCare forward for MI455X (gfx1250, wave32, WMMA).
// Dominant cost: per-feature GRU recurrence -> v_wmma_f32_16x16x32_f16 with
// Whh[f] resident in LDS. All dense linears use a generic WMMA GEMM kernel.
// Round 2: fast sigmoid/tanh via v_rcp_f32 (kills v_div_scale chains on the
// recurrence critical path), hoisted A-fragment loads (4x fewer ds loads).
// ---------------------------------------------------------------------------

#define Bn    64
#define Tn    96
#define Fn    48
#define Hn    128
#define An    8
#define DFFn  512
#define DEMOn 12
#define NHn   4
#define Nn    49          // F + 1 positions
#define DMn   128

typedef __attribute__((ext_vector_type(16))) _Float16 v16h;
typedef __attribute__((ext_vector_type(8)))  float    v8f;

// fast sigmoid: exp + add + v_rcp_f32 (no IEEE division sequence)
__device__ __forceinline__ float sigmoidf_(float v) {
  return __builtin_amdgcn_rcpf(1.0f + __expf(-v));
}
// fast tanh: tanh(x) = 1 - 2/(exp(2x)+1)
__device__ __forceinline__ float tanhf_(float v) {
  return 1.0f - 2.0f * __builtin_amdgcn_rcpf(1.0f + __expf(2.0f * v));
}

// Load a 16x32 f16 fragment (A-style / N-major-B-style) from an LDS tile.
// ISA 7.12.2 layout: lanes 0-15 row=lane, VGPR i holds K pair {2i | i<4 : 8+2i}
// lanes 16-31 same rows, K offset +8 (low VGPRs) / +24..31 (high VGPRs).
__device__ __forceinline__ v16h load_frag16(const _Float16* base, int ld, int row0, int k0) {
  const int lane  = threadIdx.x & 31;
  const int r     = row0 + (lane & 15);
  const int hioff = (lane & 16) ? 8 : 0;
  const _Float16* p = base + r * ld + k0 + hioff;
  v16h f;
#pragma unroll
  for (int i = 0; i < 8; ++i) {
    const int k = 2 * i + ((i & 4) ? 8 : 0);
    f[2 * i]     = p[k];
    f[2 * i + 1] = p[k + 1];
  }
  return f;
}

// ---------------------------------------------------------------------------
// Kernel 1: per-feature GRU scan. grid = F blocks, 256 threads (8 waves).
// Per step: gh = h[64x128] @ Whh[f][384x128]^T via WMMA, fused gate math,
// write h back to LDS f16 and stream Hs[b,t,f,:] (f16) to global.
// ---------------------------------------------------------------------------
__global__ __launch_bounds__(256) void gru_scan_kernel(
    const float* __restrict__ x,     // [B,T,F]
    const float* __restrict__ Wih,   // [F,3H]
    const float* __restrict__ Whh,   // [F,3H,H]
    const float* __restrict__ bih,   // [F,3H]
    const float* __restrict__ bhh,   // [F,3H]
    _Float16* __restrict__ Hs)       // [B,T,F,H] f16
{
  extern __shared__ char smem[];
  _Float16* Wl   = (_Float16*)smem;                 // [384][128] f16  (96 KB)
  _Float16* hl   = Wl + 384 * 128;                  // [64][128]  f16  (16 KB)
  float*    xbuf = (float*)(hl + 64 * 128);         // [64]
  float*    wih  = xbuf + 64;                       // [384]
  float*    bi   = wih + 384;                       // [384]
  float*    bh   = bi + 384;                        // [384]

  const int f    = blockIdx.x;
  const int tid  = threadIdx.x;
  const int wave = tid >> 5;
  const int lane = tid & 31;

  const float* Wg = Whh + (size_t)f * 384 * 128;
  for (int i = tid; i < 384 * 128; i += 256) Wl[i] = (_Float16)Wg[i];
  for (int i = tid; i < 384; i += 256) {
    wih[i] = Wih[f * 384 + i];
    bi[i]  = bih[f * 384 + i];
    bh[i]  = bhh[f * 384 + i];
  }
  for (int i = tid; i < 64 * 128; i += 256) hl[i] = (_Float16)0.0f;
  __syncthreads();

  const int m     = wave >> 1;            // batch tile 0..3 (rows 16m..16m+15)
  const int tbase = (wave & 1) * 4;       // column-tile group base (t = tbase..tbase+3)
  const int rlo   = lane & 15;
  const int hi8   = (lane & 16) ? 8 : 0;

  for (int step = 0; step < Tn; ++step) {
    if (tid < 64) xbuf[tid] = x[((size_t)tid * Tn + step) * Fn + f];
    __syncthreads();

    // A fragments (h tile) depend only on kk -> load once, reuse for 12 WMMAs.
    v16h afr[4];
#pragma unroll
    for (int kk = 0; kk < 4; ++kk) afr[kk] = load_frag16(hl, 128, 16 * m, 32 * kk);

    float hnew[4][8];
#pragma unroll
    for (int j = 0; j < 4; ++j) {
      const int t = tbase + j;
      v8f acc[3] = {};
#pragma unroll
      for (int kk = 0; kk < 4; ++kk) {
#pragma unroll
        for (int s = 0; s < 3; ++s) {
          v16h bf = load_frag16(Wl, 128, s * 128 + 16 * t, 32 * kk);
          acc[s] = __builtin_amdgcn_wmma_f32_16x16x32_f16(
              false, afr[kk], false, bf, (short)0, acc[s], false, false);
        }
      }
      const int col = 16 * t + rlo;       // h-dim column in [0,128)
#pragma unroll
      for (int p = 0; p < 8; ++p) {
        const int b  = 16 * m + p + hi8;
        const float xv = xbuf[b];
        const float ir = xv * wih[col]       + bi[col];
        const float iz = xv * wih[col + 128] + bi[col + 128];
        const float in = xv * wih[col + 256] + bi[col + 256];
        const float hr = acc[0][p] + bh[col];
        const float hz = acc[1][p] + bh[col + 128];
        const float hn = acc[2][p] + bh[col + 256];
        const float r  = sigmoidf_(ir + hr);
        const float z  = sigmoidf_(iz + hz);
        const float nn = tanhf_(in + r * hn);
        const float ho = (float)hl[b * 128 + col];
        hnew[j][p] = (1.0f - z) * nn + z * ho;
      }
    }
    __syncthreads();                      // all WMMA reads of hl done
#pragma unroll
    for (int j = 0; j < 4; ++j) {
      const int col = 16 * (tbase + j) + rlo;
#pragma unroll
      for (int p = 0; p < 8; ++p) {
        const int b = 16 * m + p + hi8;
        const _Float16 hv = (_Float16)hnew[j][p];
        hl[b * 128 + col] = hv;
        Hs[(((size_t)b * Tn + step) * Fn + f) * Hn + col] = hv;
      }
    }
    __syncthreads();
  }
}

// ---------------------------------------------------------------------------
// Kernel 2: generic WMMA linear: C[M,N] = act(A[M,K](lda) @ W[N,K]^T + bias)
// (+ optional residual). 64x64 tile per block, 8 waves x 2 output tiles.
// ---------------------------------------------------------------------------
__global__ __launch_bounds__(256) void wmma_linear_kernel(
    const float* __restrict__ A, int lda,
    const float* __restrict__ W,
    const float* __restrict__ bias,
    const float* __restrict__ res,
    float* __restrict__ C, int ldc,
    int M, int K, int N, int act)
{
  __shared__ _Float16 As[64][32];
  __shared__ _Float16 Bs[64][32];
  const int mb   = blockIdx.x * 64;
  const int nb   = blockIdx.y * 64;
  const int tid  = threadIdx.x;
  const int wave = tid >> 5;
  const int lane = tid & 31;

  v8f acc[2] = {};
  for (int k0 = 0; k0 < K; k0 += 32) {
    __syncthreads();
    for (int i = tid; i < 64 * 32; i += 256) {
      const int r = i >> 5, c = i & 31;
      const int gr = mb + r, gc = k0 + c;
      As[r][c] = (_Float16)((gr < M && gc < K) ? A[(size_t)gr * lda + gc] : 0.0f);
    }
    for (int i = tid; i < 64 * 32; i += 256) {
      const int r = i >> 5, c = i & 31;
      const int gr = nb + r, gc = k0 + c;
      Bs[r][c] = (_Float16)((gr < N && gc < K) ? W[(size_t)gr * K + gc] : 0.0f);
    }
    __syncthreads();
#pragma unroll
    for (int u = 0; u < 2; ++u) {
      const int tIdx = wave * 2 + u;
      const int tm = tIdx >> 2, tn = tIdx & 3;
      v16h af = load_frag16(&As[0][0], 32, tm * 16, 0);
      v16h bf = load_frag16(&Bs[0][0], 32, tn * 16, 0);
      acc[u] = __builtin_amdgcn_wmma_f32_16x16x32_f16(
          false, af, false, bf, (short)0, acc[u], false, false);
    }
  }
  const int rlo = lane & 15, hi8 = (lane & 16) ? 8 : 0;
#pragma unroll
  for (int u = 0; u < 2; ++u) {
    const int tIdx = wave * 2 + u;
    const int tm = tIdx >> 2, tn = tIdx & 3;
#pragma unroll
    for (int p = 0; p < 8; ++p) {
      const int gr = mb + tm * 16 + p + hi8;
      const int gc = nb + tn * 16 + rlo;
      if (gr < M && gc < N) {
        float v = acc[u][p] + (bias ? bias[gc] : 0.0f);
        if (act == 1)      v = fmaxf(v, 0.0f);
        else if (act == 2) v = tanhf_(v);
        if (res) v += res[(size_t)gr * ldc + gc];
        C[(size_t)gr * ldc + gc] = v;
      }
    }
  }
}

// ---------------------------------------------------------------------------
// Kernel 3: time-aware per-feature attention. grid = (B, F), 128 threads.
// ---------------------------------------------------------------------------
__global__ __launch_bounds__(128) void feat_attention_kernel(
    const _Float16* __restrict__ Hs,   // [B,T,F,H]
    const float* __restrict__ Wt,      // [F,H,A]
    const float* __restrict__ Wx,      // [F,H,A]
    const float* __restrict__ rate,    // [F]
    float* __restrict__ posi)          // [B,Nn,H] (writes rows 0..F-1)
{
  const int b = blockIdx.x, f = blockIdx.y, tid = threadIdx.x;
  __shared__ float q[An];
  __shared__ float e[Tn];
  __shared__ float aw[Tn];
  __shared__ float invs;

  const _Float16* hlast = Hs + (((size_t)b * Tn + (Tn - 1)) * Fn + f) * Hn;
  if (tid < An) {
    float s = 0.0f;
    for (int h = 0; h < Hn; ++h) s += (float)hlast[h] * Wt[(f * Hn + h) * An + tid];
    q[tid] = s;
  }
  __syncthreads();
  if (tid < Tn) {
    const _Float16* ht = Hs + (((size_t)b * Tn + tid) * Fn + f) * Hn;
    float dot = 0.0f;
    for (int a = 0; a < An; ++a) {
      float k = 0.0f;
      for (int h = 0; h < Hn; ++h) k += (float)ht[h] * Wx[(f * Hn + h) * An + a];
      dot += k * q[a];
    }
    const float sd    = sigmoidf_(dot);
    const float sr    = sigmoidf_(rate[f]);
    const float btime = (float)(Tn - tid);
    const float den   = sr * (__logf(2.72f + (1.0f - sd)) * btime);
    const float ev    = sd * __builtin_amdgcn_rcpf(den);
    e[tid] = ev > 0.0f ? ev : 0.0f;
  }
  __syncthreads();
  if (tid == 0) {
    float mx = -1e30f;
    for (int t = 0; t < Tn; ++t) mx = fmaxf(mx, e[t]);
    float s = 0.0f;
    for (int t = 0; t < Tn; ++t) { const float w = __expf(e[t] - mx); aw[t] = w; s += w; }
    invs = __builtin_amdgcn_rcpf(s);
  }
  __syncthreads();
  const float inv = invs;
  float acc = 0.0f;
  for (int t = 0; t < Tn; ++t)
    acc += aw[t] * inv * (float)Hs[(((size_t)b * Tn + t) * Fn + f) * Hn + tid];
  posi[((size_t)b * Nn + f) * Hn + tid] = acc;
}

// Demographics row: posi[b][F][h] = tanh(demo[b] . demoW[h] + demob[h])
__global__ __launch_bounds__(128) void demo_kernel(
    const float* __restrict__ demo, const float* __restrict__ dW,
    const float* __restrict__ db, float* __restrict__ posi)
{
  const int b = blockIdx.x, h = threadIdx.x;
  float s = db[h];
  for (int d = 0; d < DEMOn; ++d) s += demo[b * DEMOn + d] * dW[h * DEMOn + d];
  posi[((size_t)b * Nn + Fn) * Hn + h] = tanhf_(s);
}

// LayerNorm (custom: unbiased std, eps added to std). one row per block.
__global__ __launch_bounds__(128) void layernorm_kernel(
    const float* __restrict__ X, const float* __restrict__ ga,
    const float* __restrict__ gb, float* __restrict__ Y)
{
  __shared__ float s1[128];
  __shared__ float s2[128];
  const int r = blockIdx.x, tid = threadIdx.x;
  const float v = X[(size_t)r * DMn + tid];
  s1[tid] = v; __syncthreads();
  for (int off = 64; off > 0; off >>= 1) { if (tid < off) s1[tid] += s1[tid + off]; __syncthreads(); }
  const float mean = s1[0] / (float)DMn;
  const float d = v - mean;
  s2[tid] = d * d; __syncthreads();
  for (int off = 64; off > 0; off >>= 1) { if (tid < off) s2[tid] += s2[tid + off]; __syncthreads(); }
  const float stdv = sqrtf(s2[0] / (float)(DMn - 1));
  Y[(size_t)r * DMn + tid] = ga[tid] * d * __builtin_amdgcn_rcpf(stdv + 1e-7f) + gb[tid];
}

// MHA softmax attention per (b, head). dk = 32, N = 49.
__global__ __launch_bounds__(64) void mha_attn_kernel(
    const float* __restrict__ Q, const float* __restrict__ Km,
    const float* __restrict__ V, float* __restrict__ ctx)
{
  const int b = blockIdx.x, hd = blockIdx.y, n = threadIdx.x;
  if (n >= Nn) return;
  const int dk = DMn / NHn;
  const float* qp = Q + ((size_t)b * Nn + n) * DMn + hd * dk;
  float sc[Nn];
  float mx = -1e30f;
  for (int mI = 0; mI < Nn; ++mI) {
    const float* kp = Km + ((size_t)b * Nn + mI) * DMn + hd * dk;
    float s = 0.0f;
    for (int d = 0; d < dk; ++d) s += qp[d] * kp[d];
    s *= 0.17677669529663687f;     // 1/sqrt(32)
    sc[mI] = s; mx = fmaxf(mx, s);
  }
  float sum = 0.0f;
  for (int mI = 0; mI < Nn; ++mI) { sc[mI] = __expf(sc[mI] - mx); sum += sc[mI]; }
  const float inv = __builtin_amdgcn_rcpf(sum);
  for (int d = 0; d < dk; ++d) {
    float a = 0.0f;
    for (int mI = 0; mI < Nn; ++mI) a += sc[mI] * V[((size_t)b * Nn + mI) * DMn + hd * dk + d];
    ctx[((size_t)b * Nn + n) * DMn + hd * dk + d] = a * inv;
  }
}

__global__ __launch_bounds__(256) void add_kernel(
    const float* __restrict__ X, const float* __restrict__ Y,
    float* __restrict__ Z, int count)
{
  const int i = blockIdx.x * 256 + threadIdx.x;
  if (i < count) Z[i] = X[i] + Y[i];
}

__global__ void zero_scalar_kernel(float* p) { if (threadIdx.x == 0) p[0] = 0.0f; }

// DeCov: one block per position n; cov over batch, off-diagonal Frobenius.
__global__ __launch_bounds__(256) void decov_kernel(
    const float* __restrict__ mo, float* __restrict__ out)
{
  __shared__ float xc[Bn][DMn];      // 32 KB
  __shared__ float mean[DMn];
  __shared__ float part[256];
  const int n = blockIdx.x, tid = threadIdx.x;
  if (tid < DMn) {
    float s = 0.0f;
    for (int b = 0; b < Bn; ++b) s += mo[((size_t)b * Nn + n) * DMn + tid];
    mean[tid] = s / (float)Bn;
  }
  __syncthreads();
  for (int i = tid; i < Bn * DMn; i += 256) {
    const int b = i >> 7, h = i & 127;
    xc[b][h] = mo[((size_t)b * Nn + n) * DMn + h] - mean[h];
  }
  __syncthreads();
  float acc = 0.0f;
  for (int pi = tid; pi < DMn * DMn; pi += 256) {
    const int h = pi >> 7, g = pi & 127;
    float s = 0.0f;
    for (int b = 0; b < Bn; ++b) s += xc[b][h] * xc[b][g];
    const float c = s / (float)(Bn - 1);
    if (h != g) acc += c * c;
  }
  part[tid] = acc;
  __syncthreads();
  for (int off = 128; off > 0; off >>= 1) { if (tid < off) part[tid] += part[tid + off]; __syncthreads(); }
  if (tid == 0) atomicAdd(&out[Bn], 0.5f * part[0]);
}

// Final attention pooling + sigmoid output. one block per b.
__global__ __launch_bounds__(64) void final_out_kernel(
    const float* __restrict__ fq, const float* __restrict__ fk,
    const float* __restrict__ fv, const float* __restrict__ oW,
    const float* __restrict__ ob, float* __restrict__ out)
{
  const int b = blockIdx.x, tid = threadIdx.x;
  __shared__ float fe[Nn];
  __shared__ float pooled[DMn];
  if (tid < Nn) {
    float s = 0.0f;
    for (int a = 0; a < DMn; ++a) s += fk[((size_t)b * Nn + tid) * DMn + a] * fq[(size_t)b * DMn + a];
    fe[tid] = s;
  }
  __syncthreads();
  if (tid == 0) {
    float mx = -1e30f;
    for (int i = 0; i < Nn; ++i) mx = fmaxf(mx, fe[i]);
    float sum = 0.0f;
    for (int i = 0; i < Nn; ++i) { fe[i] = __expf(fe[i] - mx); sum += fe[i]; }
    const float inv = __builtin_amdgcn_rcpf(sum);
    for (int i = 0; i < Nn; ++i) fe[i] *= inv;
  }
  __syncthreads();
  for (int a = tid; a < DMn; a += 64) {
    float s = 0.0f;
    for (int nI = 0; nI < Nn; ++nI) s += fe[nI] * fv[((size_t)b * Nn + nI) * DMn + a];
    pooled[a] = s;
  }
  __syncthreads();
  if (tid == 0) {
    float s = ob[0];
    for (int a = 0; a < DMn; ++a) s += pooled[a] * oW[a];
    out[b] = sigmoidf_(s);
  }
}

// ---------------------------------------------------------------------------
extern "C" void kernel_launch(void* const* d_in, const int* in_sizes, int n_in,
                              void* d_out, int out_size, void* d_ws, size_t ws_size,
                              hipStream_t stream) {
  (void)in_sizes; (void)n_in; (void)out_size; (void)ws_size;
  const float* x     = (const float*)d_in[0];
  const float* demo  = (const float*)d_in[1];
  const float* gWih  = (const float*)d_in[2];
  const float* gWhh  = (const float*)d_in[3];
  const float* gbih  = (const float*)d_in[4];
  const float* gbhh  = (const float*)d_in[5];
  const float* attWt = (const float*)d_in[6];
  const float* attWx = (const float*)d_in[7];
  const float* attR  = (const float*)d_in[8];
  const float* demoW = (const float*)d_in[9];
  const float* demob = (const float*)d_in[10];
  const float* mWq = (const float*)d_in[11]; const float* mbq = (const float*)d_in[12];
  const float* mWk = (const float*)d_in[13]; const float* mbk = (const float*)d_in[14];
  const float* mWv = (const float*)d_in[15]; const float* mbv = (const float*)d_in[16];
  const float* mWo = (const float*)d_in[17]; const float* mbo = (const float*)d_in[18];
  const float* ln1a = (const float*)d_in[19]; const float* ln1b = (const float*)d_in[20];
  const float* ln2a = (const float*)d_in[21]; const float* ln2b = (const float*)d_in[22];
  const float* fW1 = (const float*)d_in[23]; const float* fb1 = (const float*)d_in[24];
  const float* fW2 = (const float*)d_in[25]; const float* fb2 = (const float*)d_in[26];
  const float* fWqW = (const float*)d_in[27]; const float* fbq = (const float*)d_in[28];
  const float* fWkW = (const float*)d_in[29]; const float* fbk = (const float*)d_in[30];
  const float* fWvW = (const float*)d_in[31]; const float* fbv = (const float*)d_in[32];
  const float* oW  = (const float*)d_in[33]; const float* ob  = (const float*)d_in[34];
  float* out = (float*)d_out;

  // workspace carve-up
  char* ws = (char*)d_ws;
  size_t off = 0;
  auto carve = [&](size_t bytes) { size_t o = off; off = (off + bytes + 255) & ~(size_t)255; return o; };
  const size_t POSI = (size_t)Bn * Nn * DMn * sizeof(float);     // 1.6 MB
  _Float16* Hs  = (_Float16*)(ws + carve((size_t)Bn * Tn * Fn * Hn * sizeof(_Float16)));
  float* posi   = (float*)(ws + carve(POSI));
  float* xn     = (float*)(ws + carve(POSI));
  float* qb     = (float*)(ws + carve(POSI));
  float* kb     = (float*)(ws + carve(POSI));
  float* vb     = (float*)(ws + carve(POSI));
  float* ctx    = (float*)(ws + carve(POSI));
  float* mo     = (float*)(ws + carve(POSI));
  float* h1     = (float*)(ws + carve(POSI));
  float* xn2    = (float*)(ws + carve(POSI));
  float* ffmid  = (float*)(ws + carve((size_t)Bn * Nn * DFFn * sizeof(float)));
  float* h2     = (float*)(ws + carve(POSI));
  float* fqv    = (float*)(ws + carve((size_t)Bn * DMn * sizeof(float)));
  float* fkv    = (float*)(ws + carve(POSI));
  float* fvv    = (float*)(ws + carve(POSI));

  const int MROWS = Bn * Nn;   // 3136

  // 1) GRU scan (dominant, WMMA). 117 KB dynamic LDS per block.
  const size_t gruLds = (size_t)(384 * 128 + 64 * 128) * sizeof(_Float16)
                      + (size_t)(64 + 3 * 384) * sizeof(float);
  (void)hipFuncSetAttribute((const void*)gru_scan_kernel,
                            hipFuncAttributeMaxDynamicSharedMemorySize, (int)gruLds);
  gru_scan_kernel<<<Fn, 256, gruLds, stream>>>(x, gWih, gWhh, gbih, gbhh, Hs);

  // 2) time-aware per-feature attention -> posi[:, 0:48, :]
  feat_attention_kernel<<<dim3(Bn, Fn), 128, 0, stream>>>(Hs, attWt, attWx, attR, posi);
  // 3) demographics row -> posi[:, 48, :]
  demo_kernel<<<Bn, 128, 0, stream>>>(demo, demoW, demob, posi);

  // 4) LN1
  layernorm_kernel<<<MROWS, 128, 0, stream>>>(posi, ln1a, ln1b, xn);

  // 5) Q/K/V projections (WMMA GEMM)
  wmma_linear_kernel<<<dim3(MROWS / 64, 2), 256, 0, stream>>>(xn, DMn, mWq, mbq, nullptr, qb, DMn, MROWS, DMn, DMn, 0);
  wmma_linear_kernel<<<dim3(MROWS / 64, 2), 256, 0, stream>>>(xn, DMn, mWk, mbk, nullptr, kb, DMn, MROWS, DMn, DMn, 0);
  wmma_linear_kernel<<<dim3(MROWS / 64, 2), 256, 0, stream>>>(xn, DMn, mWv, mbv, nullptr, vb, DMn, MROWS, DMn, DMn, 0);

  // 6) MHA softmax attention
  mha_attn_kernel<<<dim3(Bn, NHn), 64, 0, stream>>>(qb, kb, vb, ctx);

  // 7) output projection (WMMA), residual, DeCov
  wmma_linear_kernel<<<dim3(MROWS / 64, 2), 256, 0, stream>>>(ctx, DMn, mWo, mbo, nullptr, mo, DMn, MROWS, DMn, DMn, 0);
  add_kernel<<<(MROWS * DMn + 255) / 256, 256, 0, stream>>>(posi, mo, h1, MROWS * DMn);
  zero_scalar_kernel<<<1, 1, 0, stream>>>(out + Bn);
  decov_kernel<<<Nn, 256, 0, stream>>>(mo, out);

  // 8) FFN sublayer (WMMA), residual fused into second GEMM
  layernorm_kernel<<<MROWS, 128, 0, stream>>>(h1, ln2a, ln2b, xn2);
  wmma_linear_kernel<<<dim3(MROWS / 64, DFFn / 64), 256, 0, stream>>>(xn2, DMn, fW1, fb1, nullptr, ffmid, DFFn, MROWS, DMn, DFFn, 1);
  wmma_linear_kernel<<<dim3(MROWS / 64, 2), 256, 0, stream>>>(ffmid, DFFn, fW2, fb2, h1, h2, DMn, MROWS, DFFn, DMn, 0);

  // 9) final attention projections (WMMA): fq from last position rows
  wmma_linear_kernel<<<dim3(1, 2), 256, 0, stream>>>(h2 + (size_t)(Nn - 1) * DMn, Nn * DMn, fWqW, fbq, nullptr, fqv, DMn, Bn, DMn, DMn, 0);
  wmma_linear_kernel<<<dim3(MROWS / 64, 2), 256, 0, stream>>>(h2, DMn, fWkW, fbk, nullptr, fkv, DMn, MROWS, DMn, DMn, 0);
  wmma_linear_kernel<<<dim3(MROWS / 64, 2), 256, 0, stream>>>(h2, DMn, fWvW, fbv, nullptr, fvv, DMn, MROWS, DMn, DMn, 0);

  // 10) pooled attention + sigmoid output -> out[0..63]
  final_out_kernel<<<Bn, 64, 0, stream>>>(fqv, fkv, fvv, oW, ob, out);
}
